// MaskedChannelAutoencoder_77506979823788
// MI455X (gfx1250) — matrix operand order repaired and checked by reading, not compiled, not run on hardware
//
#include <hip/hip_runtime.h>
#include <hip/hip_bf16.h>

typedef __attribute__((ext_vector_type(16))) _Float16 v16h;
typedef __attribute__((ext_vector_type(8)))  _Float16 v8h;
typedef __attribute__((ext_vector_type(8)))  float    v8f;
typedef __attribute__((ext_vector_type(4)))  unsigned int u32x4;
typedef __attribute__((ext_vector_type(8)))  int          i32x8;
typedef __attribute__((ext_vector_type(4)))  int          i32x4;

// TDM path: probe-confirmed builtins on both toolchains; arity differs (clang-22: 5
// args, clang-23: 6 args). Guarded so host pass / other targets fall back cleanly.
#if defined(__HIP_DEVICE_COMPILE__) && defined(__gfx1250__) && \
    __has_builtin(__builtin_amdgcn_tensor_load_to_lds) && \
    __has_builtin(__builtin_amdgcn_s_wait_tensorcnt)
#define USE_TDM 1
#else
#define USE_TDM 0
#endif

__device__ __forceinline__ float gelu_exact(float x) {
    return 0.5f * x * (1.0f + erff(x * 0.7071067811865475f));
}

// ---------------------------------------------------------------------------
// Tiled WMMA GEMM: C[M,N] = epilogue(A[M,K] @ B[K,N] + bias) (+ residual)
// Block tile 64x128, K-step 32, 256 threads (8 waves).
// Each wave owns a 16x64 output strip: one A fragment reused by 4 WMMAs.
// A panel is DMA'd by the Tensor Data Mover into a double-buffered LDS stage
// (f32), converted to f16 during fragment build; B panel is staged transposed
// ([n][k]) so B fragments are contiguous 32B LDS loads.
// M % 64 == 0, N % 128 == 0, K % 32 == 0 (true for every GEMM in this model).
// ---------------------------------------------------------------------------
__global__ __launch_bounds__(256)
void gemm_wmma_kernel(const float* __restrict__ A, const float* __restrict__ B,
                      const float* __restrict__ bias, const float* __restrict__ residual,
                      float* __restrict__ C, int M, int N, int K, int apply_gelu)
{
    __shared__ float    Af[2][64][32];    // A stage (f32), double buffered: 16 KB
    __shared__ _Float16 Bs[2][128][32];   // B stage (f16, [n][k]), double buffered: 16 KB

    const int bm   = blockIdx.y * 64;
    const int bn   = blockIdx.x * 128;
    const int tid  = threadIdx.x;
    const int wave = tid >> 5;
    const int lane = tid & 31;
    const int tr   = wave >> 1;          // 0..3 : 16-row tile within block
    const int tcg  = (wave & 1) * 4;     // 0 or 4: first of four 16-col tiles
    const int h    = lane >> 4;          // lane half
    const int l15  = lane & 15;

    v8f acc[4] = { v8f{}, v8f{}, v8f{}, v8f{} };

    auto stageA = [&](int k0, int buf) {
#if USE_TDM
        if (wave == 0) {
            // Build Tensor DMA descriptor (D#) per CDNA5 ISA ch.8:
            // 2D tile: 32 cols (contiguous, f32) x 64 rows, row stride K elems.
            unsigned long long ga = (unsigned long long)(A + (size_t)bm * K + k0);
            unsigned int lo = (unsigned int)(unsigned long long)&Af[buf][0][0];
            u32x4 g0 = { 1u,                                    // count=1, user mode
                         lo,                                    // lds_addr (bytes)
                         (unsigned int)ga,                      // global_addr[31:0]
                         (unsigned int)((ga >> 32) & 0x1ffffffull) | (2u << 30) }; // [56:32] | type=2
            unsigned int td0 = (unsigned int)K;   // tensor_dim0 (elems per row)
            unsigned int td1 = 64u;               // tensor_dim1 (rows)
            i32x8 g1;
            g1[0] = (int)(2u << 16);              // data_size=2 -> 4-byte elems; mask/flags 0
            g1[1] = (int)((td0 & 0xffffu) << 16);                                 // td0[15:0]
            g1[2] = (int)(((td0 >> 16) & 0xffffu) | ((td1 & 0xffffu) << 16));     // td0[31:16] | td1[15:0]
            g1[3] = (int)(((td1 >> 16) & 0xffffu) | (32u << 16));                 // td1[31:16] | tile_dim0=32
            g1[4] = 64;                           // tile_dim1 = 64 (tile_dim2 = 0)
            g1[5] = (int)K;                       // tensor_dim0_stride[31:0]
            g1[6] = 0;                            // stride[47:32] | t1_stride lo
            g1[7] = 0;
            i32x4 gz = { 0, 0, 0, 0 };
#if __clang_major__ >= 23
            i32x8 gz8 = { 0, 0, 0, 0, 0, 0, 0, 0 };
            __builtin_amdgcn_tensor_load_to_lds(g0, g1, gz, gz, gz8, 0);
#else
            __builtin_amdgcn_tensor_load_to_lds(g0, g1, gz, gz, 0);
#endif
        }
#else
        int r = tid >> 2;
        int c = (tid & 3) << 3;
        const float* src = A + (size_t)(bm + r) * K + k0 + c;
#pragma unroll
        for (int i = 0; i < 8; i++) Af[buf][r][c + i] = src[i];
#endif
    };

    auto stageB = [&](int k0, int buf) {
        int n  = tid & 127;
        int kc = (tid >> 7) << 4;   // 0 or 16
        const float* src = B + (size_t)(k0 + kc) * N + bn + n;
#pragma unroll
        for (int i = 0; i < 16; i++)
            Bs[buf][n][kc + i] = (_Float16)src[(size_t)i * N];
    };

    auto compute = [&](int buf) {
        // A fragment: lane holds row (tr*16 + l15); K run [h*16, +16)
        const float* ar = &Af[buf][tr * 16 + l15][0];
        v16h afrag;
#pragma unroll
        for (int j = 0; j < 8; j++) afrag[j]     = (_Float16)ar[h * 16 + j];
#pragma unroll
        for (int j = 0; j < 8; j++) afrag[8 + j] = (_Float16)ar[h * 16 + 8 + j];
#pragma unroll
        for (int tc = 0; tc < 4; tc++) {
            // B fragment: lane holds col l15 of tile, K run [h*16, +16)
            v16h bfrag = *(const v16h*)&Bs[buf][(tcg + tc) * 16 + l15][h * 16];
            acc[tc] = __builtin_amdgcn_wmma_f32_16x16x32_f16(false, afrag, false, bfrag,
                                                             (short)0, acc[tc], false, false);
        }
    };

    const int nT = K >> 5;
    stageA(0, 0);
    stageB(0, 0);
#if USE_TDM
    __builtin_amdgcn_s_wait_tensorcnt((short)0);
#endif
    __syncthreads();

    for (int i = 0; i < nT; ++i) {
        int cur = i & 1, nxt = cur ^ 1;
        if (i + 1 < nT) {               // prefetch next tile while computing this one
            stageA((i + 1) << 5, nxt);  // TDM DMA (async, TENSORcnt) or manual fallback
            stageB((i + 1) << 5, nxt);
        }
        compute(cur);
#if USE_TDM
        __builtin_amdgcn_s_wait_tensorcnt((short)0);
#endif
        __syncthreads();
    }

    // epilogue: m = r + 8*h within the 16-row tile, n = l15 within each 16-col tile
    const int mb = bm + tr * 16 + h * 8;
#pragma unroll
    for (int tc = 0; tc < 4; tc++) {
        int n = bn + (tcg + tc) * 16 + l15;
        float bv = bias ? bias[n] : 0.0f;
#pragma unroll
        for (int r = 0; r < 8; r++) {
            int m = mb + r;
            float v = acc[tc][r] + bv;
            if (apply_gelu) v = gelu_exact(v);
            if (residual)   v += residual[(size_t)m * N + n];
            C[(size_t)m * N + n] = v;
        }
    }
}

// ---------------------------------------------------------------------------
// Direct Conv1d: x(N,Cin,Lin) -> y(N,Cout,Lout), weight (Cout,Cin,K)
// ---------------------------------------------------------------------------
__global__ __launch_bounds__(256)
void conv1d_kernel(const float* __restrict__ x, const float* __restrict__ w,
                   const float* __restrict__ bias, float* __restrict__ y,
                   int Cin, int Lin, int Cout, int Lout,
                   int Kk, int stride, int pad, int apply_gelu, long total)
{
    long idx = (long)blockIdx.x * blockDim.x + threadIdx.x;
    if (idx >= total) return;
    int  t = (int)(idx % Lout);
    long r = idx / Lout;
    int  o = (int)(r % Cout);
    long n = r / Cout;

    float acc = bias[o];
    for (int j = 0; j < Kk; j++) {
        int xi = t * stride + j - pad;
        if (xi >= 0 && xi < Lin) {
            const float* xp = x + ((size_t)n * Cin) * Lin + xi;
            const float* wp = w + ((size_t)o * Cin) * Kk + j;
            for (int i = 0; i < Cin; i++)
                acc += wp[(size_t)i * Kk] * xp[(size_t)i * Lin];
        }
    }
    if (apply_gelu) acc = gelu_exact(acc);
    y[idx] = acc;
}

// ---------------------------------------------------------------------------
// ConvTranspose1d (stride 2) via lhs-dilation math: lo = k-1-p
// weight layout (Cout,Cin,K) matching the reference's 'OIH'
// ---------------------------------------------------------------------------
__global__ __launch_bounds__(256)
void deconv1d_kernel(const float* __restrict__ x, const float* __restrict__ w,
                     const float* __restrict__ bias, float* __restrict__ y,
                     int Cin, int Lin, int Cout, int Lout,
                     int Kk, int lo, int apply_gelu, long total)
{
    long idx = (long)blockIdx.x * blockDim.x + threadIdx.x;
    if (idx >= total) return;
    int  t = (int)(idx % Lout);
    long r = idx / Lout;
    int  o = (int)(r % Cout);
    long n = r / Cout;

    int Ld = 2 * Lin - 1;  // dilated length
    float acc = bias[o];
    for (int j = 0; j < Kk; j++) {
        int u = t + j - lo;
        if (u >= 0 && u < Ld && (u & 1) == 0) {
            int xi = u >> 1;
            const float* xp = x + ((size_t)n * Cin) * Lin + xi;
            const float* wp = w + ((size_t)o * Cin) * Kk + j;
            for (int i = 0; i < Cin; i++)
                acc += wp[(size_t)i * Kk] * xp[(size_t)i * Lin];
        }
    }
    if (apply_gelu) acc = gelu_exact(acc);
    y[idx] = acc;
}

// ---------------------------------------------------------------------------
// Mean-pool over T=Lt then mask-token select: x[n,d] = mask[n]? tok[d] : mean
// a3 layout (NC, D, Lt)
// ---------------------------------------------------------------------------
__global__ __launch_bounds__(256)
void pool_mask_kernel(const float* __restrict__ a3, const unsigned char* __restrict__ mask,
                      const float* __restrict__ tok, float* __restrict__ x,
                      int NC, int D, int Lt)
{
    long idx = (long)blockIdx.x * blockDim.x + threadIdx.x;
    if (idx >= (long)NC * D) return;
    int n = (int)(idx / D);
    int d = (int)(idx % D);
    float s = 0.f;
    const float* p = a3 + ((size_t)n * D + d) * Lt;
    for (int t = 0; t < Lt; t++) s += p[t];
    s /= (float)Lt;
    x[idx] = mask[n] ? tok[d] : s;
}

// ---------------------------------------------------------------------------
// LayerNorm over D=128 (population variance, eps 1e-5). In-place safe.
// ---------------------------------------------------------------------------
__global__ __launch_bounds__(128)
void ln_kernel(const float* __restrict__ x, const float* __restrict__ g,
               const float* __restrict__ b, float* __restrict__ y)
{
    __shared__ float red[128];
    int row = blockIdx.x, t = threadIdx.x;
    float v = x[(size_t)row * 128 + t];
    red[t] = v; __syncthreads();
#pragma unroll
    for (int s = 64; s > 0; s >>= 1) { if (t < s) red[t] += red[t + s]; __syncthreads(); }
    float mean = red[0] * (1.f / 128.f); __syncthreads();
    float d = v - mean;
    red[t] = d * d; __syncthreads();
#pragma unroll
    for (int s = 64; s > 0; s >>= 1) { if (t < s) red[t] += red[t + s]; __syncthreads(); }
    float var = red[0] * (1.f / 128.f);
    y[(size_t)row * 128 + t] = d * rsqrtf(var + 1e-5f) * g[t] + b[t];
}

// ---------------------------------------------------------------------------
// Attention per (batch, head): 96 queries, 96 keys, head_dim 32, online softmax
// qkv rows: [0:128) q | [128:256) k | [256:384) v
// ---------------------------------------------------------------------------
__global__ __launch_bounds__(96)
void attn_kernel(const float* __restrict__ qkv, float* __restrict__ att)
{
    __shared__ float Ks[96][32];
    __shared__ float Vs[96][32];
    int bh = blockIdx.x;
    int b  = bh >> 2;
    int h  = bh & 3;
    int t  = threadIdx.x;
    const float* base = qkv + (size_t)b * 96 * 384;

    for (int idx = t; idx < 96 * 32; idx += 96) {
        int j = idx >> 5, d = idx & 31;
        Ks[j][d] = base[(size_t)j * 384 + 128 + h * 32 + d];
        Vs[j][d] = base[(size_t)j * 384 + 256 + h * 32 + d];
    }
    __syncthreads();

    float q[32];
#pragma unroll
    for (int d = 0; d < 32; d++) q[d] = base[(size_t)t * 384 + h * 32 + d];

    float m = -1e30f, l = 0.f, acc[32];
#pragma unroll
    for (int d = 0; d < 32; d++) acc[d] = 0.f;

    for (int j = 0; j < 96; j++) {
        float s = 0.f;
#pragma unroll
        for (int d = 0; d < 32; d++) s += q[d] * Ks[j][d];
        s *= 0.17677669529663687f;  // 1/sqrt(32)
        float mn   = fmaxf(m, s);
        float corr = __expf(m - mn);
        float p    = __expf(s - mn);
        l = l * corr + p;
#pragma unroll
        for (int d = 0; d < 32; d++) acc[d] = acc[d] * corr + p * Vs[j][d];
        m = mn;
    }
    float inv = 1.f / l;
    float* o = att + ((size_t)b * 96 + t) * 128 + h * 32;
#pragma unroll
    for (int d = 0; d < 32; d++) o[d] = acc[d] * inv;
}

// ---------------------------------------------------------------------------
// Loss: masked L1
// ---------------------------------------------------------------------------
__global__ void zero_acc_kernel(float* p) { if (threadIdx.x < 2) p[threadIdx.x] = 0.f; }

__global__ __launch_bounds__(256)
void loss_accum_kernel(const float* __restrict__ pred, const float* __restrict__ data,
                       const unsigned char* __restrict__ mask, const int* __restrict__ counts,
                       float* __restrict__ acc, int Cm, int T)
{
    __shared__ float sh[256];
    int bc = blockIdx.x;
    int b  = bc / Cm;
    int c  = bc % Cm;
    bool lm = (mask[bc] != 0) && (c < counts[b]);
    int t = threadIdx.x;
    float s = 0.f;
    if (lm) {
        const float* pp = pred + (size_t)bc * T;
        const float* dp = data + (size_t)bc * T;
        for (int i = t; i < T; i += 256) s += fabsf(pp[i] - dp[i]);
    }
    sh[t] = s; __syncthreads();
#pragma unroll
    for (int k = 128; k > 0; k >>= 1) { if (t < k) sh[t] += sh[t + k]; __syncthreads(); }
    if (t == 0 && lm) {
        atomicAdd(&acc[0], sh[0]);
        atomicAdd(&acc[1], 1.0f);
    }
}

__global__ void loss_final_kernel(const float* __restrict__ acc, float* __restrict__ out, int T)
{
    float n = fmaxf(acc[1] * (float)T, 1.0f);
    out[0] = acc[0] / n;
}

// ---------------------------------------------------------------------------
// Orchestration
// ---------------------------------------------------------------------------
static inline int cdiv_l(long a, long b) { return (int)((a + b - 1) / b); }

extern "C" void kernel_launch(void* const* d_in, const int* in_sizes, int n_in,
                              void* d_out, int out_size, void* d_ws, size_t ws_size,
                              hipStream_t stream)
{
    const int Bb = 128, Cm = 96, T = 256, D = 128, NL = 4, FFd = 512;
    const int NC = Bb * Cm;  // 12288

    // Input order: data, then params pytree leaves (sorted dict keys), counts, mask
    const float* data = (const float*)d_in[0];
    const float* dec_w[3] = {(const float*)d_in[1], (const float*)d_in[3], (const float*)d_in[5]};
    const float* dec_b[3] = {(const float*)d_in[2], (const float*)d_in[4], (const float*)d_in[6]};
    const float* enc_w[3] = {(const float*)d_in[7], (const float*)d_in[9], (const float*)d_in[11]};
    const float* enc_b[3] = {(const float*)d_in[8], (const float*)d_in[10], (const float*)d_in[12]};
    const float* mask_token = (const float*)d_in[61];
    const float* norm_g = (const float*)d_in[62];
    const float* norm_b = (const float*)d_in[63];
    const float* proj_w = (const float*)d_in[64];
    const float* proj_b = (const float*)d_in[65];
    const int*   counts = (const int*)d_in[66];
    const unsigned char* mask = (const unsigned char*)d_in[67];

    float* ws = (float*)d_ws;
    const size_t BIG = (size_t)NC * 32 * 128;       // 50,331,648 floats
    float* Pb   = ws;
    float* Qb   = ws + BIG;
    float* Xb   = ws + 2 * BIG;
    float* LNb  = Xb  + (size_t)NC * D;
    float* QKVb = LNb + (size_t)NC * D;
    float* ATTb = QKVb + (size_t)NC * 384;
    float* FFb  = ATTb + (size_t)NC * D;
    float* ACC  = FFb  + (size_t)NC * FFd;

    float* out  = (float*)d_out;
    float* pred = out + 1;

    // ---- encoder ----
    { long tot = (long)NC * 32 * 128;
      conv1d_kernel<<<cdiv_l(tot, 256), 256, 0, stream>>>(data, enc_w[0], enc_b[0], Pb,
          1, 256, 32, 128, 7, 2, 3, 1, tot); }
    { long tot = (long)NC * 64 * 64;
      conv1d_kernel<<<cdiv_l(tot, 256), 256, 0, stream>>>(Pb, enc_w[1], enc_b[1], Qb,
          32, 128, 64, 64, 5, 2, 2, 1, tot); }
    { long tot = (long)NC * 128 * 32;
      conv1d_kernel<<<cdiv_l(tot, 256), 256, 0, stream>>>(Qb, enc_w[2], enc_b[2], Pb,
          64, 64, 128, 32, 5, 2, 2, 1, tot); }
    pool_mask_kernel<<<cdiv_l((long)NC * D, 256), 256, 0, stream>>>(Pb, mask, mask_token, Xb,
        NC, D, 32);

    // ---- transformer layers ----
    for (int L = 0; L < NL; L++) {
        void* const* lp = d_in + 13 + L * 12;
        const float* b1    = (const float*)lp[0];
        const float* b2    = (const float*)lp[1];
        const float* b_o   = (const float*)lp[2];
        const float* b_qkv = (const float*)lp[3];
        const float* ln1g  = (const float*)lp[4];
        const float* ln1b  = (const float*)lp[5];
        const float* ln2g  = (const float*)lp[6];
        const float* ln2b  = (const float*)lp[7];
        const float* w1    = (const float*)lp[8];
        const float* w2    = (const float*)lp[9];
        const float* w_o   = (const float*)lp[10];
        const float* w_qkv = (const float*)lp[11];

        ln_kernel<<<NC, 128, 0, stream>>>(Xb, ln1g, ln1b, LNb);
        gemm_wmma_kernel<<<dim3(384 / 128, NC / 64), 256, 0, stream>>>(
            LNb, w_qkv, b_qkv, nullptr, QKVb, NC, 384, 128, 0);
        attn_kernel<<<Bb * 4, 96, 0, stream>>>(QKVb, ATTb);
        gemm_wmma_kernel<<<dim3(128 / 128, NC / 64), 256, 0, stream>>>(
            ATTb, w_o, b_o, Xb, Xb, NC, 128, 128, 0);     // residual add, in-place safe
        ln_kernel<<<NC, 128, 0, stream>>>(Xb, ln2g, ln2b, LNb);
        gemm_wmma_kernel<<<dim3(512 / 128, NC / 64), 256, 0, stream>>>(
            LNb, w1, b1, nullptr, FFb, NC, 512, 128, 1);  // GELU
        gemm_wmma_kernel<<<dim3(128 / 128, NC / 64), 256, 0, stream>>>(
            FFb, w2, b2, Xb, Xb, NC, 128, 512, 0);        // residual add
    }
    ln_kernel<<<NC, 128, 0, stream>>>(Xb, norm_g, norm_b, Xb);  // final norm (in place)

    // ---- decoder ----
    gemm_wmma_kernel<<<dim3(4096 / 128, NC / 64), 256, 0, stream>>>(
        Xb, proj_w, proj_b, nullptr, Pb, NC, 4096, 128, 1);     // GELU; (NC,128,32) view
    { long tot = (long)NC * 64 * 64;
      deconv1d_kernel<<<cdiv_l(tot, 256), 256, 0, stream>>>(Pb, dec_w[0], dec_b[0], Qb,
          128, 32, 64, 64, 5, /*lo=*/2, 1, tot); }
    { long tot = (long)NC * 32 * 128;
      deconv1d_kernel<<<cdiv_l(tot, 256), 256, 0, stream>>>(Qb, dec_w[1], dec_b[1], Pb,
          64, 64, 32, 128, 5, /*lo=*/2, 1, tot); }
    { long tot = (long)NC * 1 * 256;
      deconv1d_kernel<<<cdiv_l(tot, 256), 256, 0, stream>>>(Pb, dec_w[2], dec_b[2], pred,
          32, 128, 1, 256, 7, /*lo=*/3, 0, tot); }

    // ---- loss ----
    zero_acc_kernel<<<1, 32, 0, stream>>>(ACC);
    loss_accum_kernel<<<Bb * Cm, 256, 0, stream>>>(pred, data, mask, counts, ACC, Cm, T);
    loss_final_kernel<<<1, 1, 0, stream>>>(ACC, out, T);
}
